// ContrastiveLoss_74526272520755
// MI455X (gfx1250) — compile-verified, matrix operand order checked by reading, not run on hardware
//
#include <hip/hip_runtime.h>
#include <hip/hip_bf16.h>

typedef __attribute__((ext_vector_type(16))) __bf16 v16bf;
typedef __attribute__((ext_vector_type(8)))  __bf16 v8bf;
typedef __attribute__((ext_vector_type(4)))  __bf16 v4bf;
typedef __attribute__((ext_vector_type(8)))  float  v8f;

#define B_DIM 2048
#define D_DIM 4096
#define C_DIM 1000
#define MT    32                      // macro-tile edge per wave
#define NMT   (B_DIM / MT)            // 64 macro-tiles per edge
#define NTRI  (NMT * (NMT + 1) / 2)   // 2080 upper-tri macro-tiles
#define GEMM_BLOCKS (NTRI / 8)        // 260 blocks x 8 waves

// ---------------- workspace layout (bytes) ----------------
#define XN_OFF   0
#define XN_BYTES ((size_t)B_DIM * D_DIM * 2)          // 16 MiB bf16
#define CE_OFF   (XN_OFF + XN_BYTES)                  // 2048 floats
#define POS_OFF  (CE_OFF + B_DIM * 4)                 // 260 floats
#define CNT_OFF  (POS_OFF + GEMM_BLOCKS * 4)          // 260 floats
#define NEG_OFF  (CNT_OFF + GEMM_BLOCKS * 4)          // 16 floats

// ---------------- pass 1: row-normalize xs -> bf16 xn ----------------
__global__ void cl_normalize(const float* __restrict__ xs, __bf16* __restrict__ xn) {
    __shared__ float red[256];
    const int row = blockIdx.x, t = threadIdx.x;
    const float4* p = reinterpret_cast<const float4*>(xs + (size_t)row * D_DIM);
    float s = 0.f;
#pragma unroll
    for (int k = 0; k < 4; ++k) {
        float4 q = p[t + k * 256];
        s += q.x * q.x + q.y * q.y + q.z * q.z + q.w * q.w;
    }
    red[t] = s; __syncthreads();
    for (int st = 128; st > 0; st >>= 1) {
        if (t < st) red[t] += red[t + st];
        __syncthreads();
    }
    const float inv = 1.f / fmaxf(sqrtf(red[0]), 1e-8f);
    __bf16* orow = xn + (size_t)row * D_DIM;
#pragma unroll
    for (int k = 0; k < 4; ++k) {
        float4 q = p[t + k * 256];
        v4bf o;
        o.x = (__bf16)(q.x * inv); o.y = (__bf16)(q.y * inv);
        o.z = (__bf16)(q.z * inv); o.w = (__bf16)(q.w * inv);
        *reinterpret_cast<v4bf*>(orow + (size_t)(t + k * 256) * 4) = o;
    }
}

// ---------------- pass 2: per-row cross entropy ----------------
__global__ void cl_ce(const float* __restrict__ yp, const int* __restrict__ yt,
                      float* __restrict__ ce_row) {
    __shared__ float red[256];
    const int row = blockIdx.x, t = threadIdx.x;
    const float* p = yp + (size_t)row * C_DIM;
    float m = -INFINITY;
    for (int c = t; c < C_DIM; c += 256) m = fmaxf(m, p[c]);
    red[t] = m; __syncthreads();
    for (int st = 128; st > 0; st >>= 1) {
        if (t < st) red[t] = fmaxf(red[t], red[t + st]);
        __syncthreads();
    }
    m = red[0]; __syncthreads();
    float s = 0.f;
    for (int c = t; c < C_DIM; c += 256) s += expf(p[c] - m);
    red[t] = s; __syncthreads();
    for (int st = 128; st > 0; st >>= 1) {
        if (t < st) red[t] += red[t + st];
        __syncthreads();
    }
    if (t == 0) ce_row[row] = (logf(red[0]) + m) - p[yt[row]];
}

// ---------------- pass 3: fused WMMA Gram-matrix + masked pos reduction ----------------
__device__ __forceinline__ v16bf load_frag(const __bf16* p) {
    // 16-bit A/B fragment: two contiguous 8-element (16B) chunks, 16B aligned.
    v8bf lo = *reinterpret_cast<const v8bf*>(p);
    v8bf hi = *reinterpret_cast<const v8bf*>(p + 16);
    return __builtin_shufflevector(lo, hi, 0, 1, 2, 3, 4, 5, 6, 7,
                                   8, 9, 10, 11, 12, 13, 14, 15);
}

__global__ void cl_gram_pos(const __bf16* __restrict__ xn, const int* __restrict__ y,
                            float* __restrict__ pos_partial, float* __restrict__ cnt_partial) {
    __shared__ int   sy[B_DIM];
    __shared__ float red[256];
    for (int i = threadIdx.x; i < B_DIM; i += 256) sy[i] = y[i];
    __syncthreads();

    const int wave = threadIdx.x >> 5;
    const int lane = threadIdx.x & 31;
    const int half = lane >> 4;
    const int l16  = lane & 15;

    // decode linear upper-tri macro-tile index -> (ti, tj), ti <= tj (wave-uniform)
    int rem = blockIdx.x * 8 + wave, ti = 0;
    while (rem >= (NMT - ti)) { rem -= (NMT - ti); ++ti; }
    const int tj = ti + rem;
    const int i0 = ti * MT, j0 = tj * MT;

    const __bf16* a0p = xn + (size_t)(i0 + l16) * D_DIM + half * 8;
    const __bf16* a1p = a0p + (size_t)16 * D_DIM;
    const __bf16* b0p = xn + (size_t)(j0 + l16) * D_DIM + half * 8;
    const __bf16* b1p = b0p + (size_t)16 * D_DIM;

    v8f c00 = {}, c01 = {}, c10 = {}, c11 = {};
#pragma unroll 4
    for (int kb = 0; kb < D_DIM; kb += 32) {
        v16bf a0 = load_frag(a0p + kb);
        v16bf a1 = load_frag(a1p + kb);
        v16bf b0 = load_frag(b0p + kb);
        v16bf b1 = load_frag(b1p + kb);
        c00 = __builtin_amdgcn_wmma_f32_16x16x32_bf16(false, a0, false, b0, (short)0, c00, false, false);
        c01 = __builtin_amdgcn_wmma_f32_16x16x32_bf16(false, a0, false, b1, (short)0, c01, false, false);
        c10 = __builtin_amdgcn_wmma_f32_16x16x32_bf16(false, a1, false, b0, (short)0, c10, false, false);
        c11 = __builtin_amdgcn_wmma_f32_16x16x32_bf16(false, a1, false, b1, (short)0, c11, false, false);
    }

    // masked accumulate: C/D layout -> lane 0-15: M=r, lane 16-31: M=r+8; N = lane&15
    float acc = 0.f, cnt = 0.f;
    const int ja = j0 + l16, jb = j0 + 16 + l16;
    const int yj0 = sy[ja], yj1 = sy[jb];
#pragma unroll
    for (int r = 0; r < 8; ++r) {
        const int m  = r + 8 * half;
        const int ia = i0 + m, ib = i0 + 16 + m;
        const int yia = sy[ia], yib = sy[ib];
        if (yia == yj0 && ia < ja) { acc += 1.f - c00[r]; cnt += 1.f; }
        if (yia == yj1 && ia < jb) { acc += 1.f - c01[r]; cnt += 1.f; }
        if (yib == yj0 && ib < ja) { acc += 1.f - c10[r]; cnt += 1.f; }
        if (yib == yj1 && ib < jb) { acc += 1.f - c11[r]; cnt += 1.f; }
    }

    red[threadIdx.x] = acc; __syncthreads();
    for (int st = 128; st > 0; st >>= 1) {
        if (threadIdx.x < st) red[threadIdx.x] += red[threadIdx.x + st];
        __syncthreads();
    }
    if (threadIdx.x == 0) pos_partial[blockIdx.x] = red[0];
    __syncthreads();
    red[threadIdx.x] = cnt; __syncthreads();
    for (int st = 128; st > 0; st >>= 1) {
        if (threadIdx.x < st) red[threadIdx.x] += red[threadIdx.x + st];
        __syncthreads();
    }
    if (threadIdx.x == 0) cnt_partial[blockIdx.x] = red[0];
}

// ---------------- pass 4: sim[0, 1..16] (first 16 pair combos) ----------------
__global__ void cl_neg(const __bf16* __restrict__ xn, float* __restrict__ neg_out) {
    const int wave = threadIdx.x >> 5, lane = threadIdx.x & 31;
    const __bf16* x0 = xn;
#pragma unroll
    for (int p = 0; p < 2; ++p) {
        const int j = wave * 2 + 1 + p;                 // j = 1..16
        const __bf16* xj = xn + (size_t)j * D_DIM;
        float dot = 0.f;
        for (int k = lane; k < D_DIM; k += 32)
            dot += (float)x0[k] * (float)xj[k];
        for (int off = 16; off > 0; off >>= 1)
            dot += __shfl_down(dot, off, 32);
        if (lane == 0) neg_out[j - 1] = fmaxf(dot, 0.f);
    }
}

// ---------------- pass 5: deterministic combine ----------------
__global__ void cl_combine(const float* __restrict__ ce_row, const float* __restrict__ pos_p,
                           const float* __restrict__ cnt_p, const float* __restrict__ neg,
                           float* __restrict__ out) {
    __shared__ float red[256];
    const int t = threadIdx.x;
    float s = 0.f;
    for (int i = t; i < B_DIM; i += 256) s += ce_row[i];
    red[t] = s; __syncthreads();
    for (int st = 128; st > 0; st >>= 1) { if (t < st) red[t] += red[t + st]; __syncthreads(); }
    const float ce_mean = red[0] / (float)B_DIM; __syncthreads();

    s = 0.f;
    for (int i = t; i < GEMM_BLOCKS; i += 256) s += pos_p[i];
    red[t] = s; __syncthreads();
    for (int st = 128; st > 0; st >>= 1) { if (t < st) red[t] += red[t + st]; __syncthreads(); }
    const float pos_sum = red[0]; __syncthreads();

    s = 0.f;
    for (int i = t; i < GEMM_BLOCKS; i += 256) s += cnt_p[i];
    red[t] = s; __syncthreads();
    for (int st = 128; st > 0; st >>= 1) { if (t < st) red[t] += red[t + st]; __syncthreads(); }
    const float cnt = red[0]; __syncthreads();

    red[t] = (t < 16) ? neg[t] : 0.f; __syncthreads();
    for (int st = 128; st > 0; st >>= 1) { if (t < st) red[t] += red[t + st]; __syncthreads(); }
    if (t == 0) {
        const float loss_pos = (cnt > 0.f) ? pos_sum / fmaxf(cnt, 1.f) : 0.f;
        out[0] = ce_mean + loss_pos + red[0] / 16.f;
    }
}

extern "C" void kernel_launch(void* const* d_in, const int* in_sizes, int n_in,
                              void* d_out, int out_size, void* d_ws, size_t ws_size,
                              hipStream_t stream) {
    const float* xs      = (const float*)d_in[0];   // [2048, 4096]
    const float* y_preds = (const float*)d_in[1];   // [2048, 1000]
    const int*   y_true  = (const int*)d_in[2];     // [2048]
    float* out = (float*)d_out;

    char* ws = (char*)d_ws;
    __bf16* xn        = (__bf16*)(ws + XN_OFF);
    float*  ce_row    = (float*)(ws + CE_OFF);
    float*  pos_part  = (float*)(ws + POS_OFF);
    float*  cnt_part  = (float*)(ws + CNT_OFF);
    float*  neg_vals  = (float*)(ws + NEG_OFF);

    cl_normalize<<<B_DIM, 256, 0, stream>>>(xs, xn);
    cl_ce<<<B_DIM, 256, 0, stream>>>(y_preds, y_true, ce_row);
    cl_gram_pos<<<GEMM_BLOCKS, 256, 0, stream>>>(xn, y_true, pos_part, cnt_part);
    cl_neg<<<1, 256, 0, stream>>>(xn, neg_vals);
    cl_combine<<<1, 256, 0, stream>>>(ce_row, pos_part, cnt_part, neg_vals, out);
}
